// ImportanceRenderer_61632780698147
// MI455X (gfx1250) — matrix operand order repaired and checked by reading, not compiled, not run on hardware
//
#include <hip/hip_runtime.h>
#include <hip/hip_bf16.h>
#include <math.h>

typedef __attribute__((ext_vector_type(16))) _Float16 v16h;
typedef __attribute__((ext_vector_type(8)))  float    v8f;
typedef __attribute__((ext_vector_type(4)))  unsigned u32x4;
typedef __attribute__((ext_vector_type(8)))  int      i32x8;
typedef __attribute__((ext_vector_type(4)))  int      i32x4;

#define Bsz   2
#define Rrays 2048
#define Ssmp  48
#define NPTS  (Bsz*Rrays*Ssmp)      /* 196608 points per level */
#define NRAYS (Bsz*Rrays)           /* 4096 */
#define HRES  256
#define PLANE_PIX (HRES*HRES)

/* d_out layout (floats) */
#define OUT_RGB   0
#define OUT_DEPTH (OUT_RGB + NRAYS*32)
#define OUT_WSUM  (OUT_DEPTH + NRAYS)
#define OUT_GRADS (OUT_WSUM + NRAYS)
#define OUT_GC    (OUT_GRADS + NRAYS*3)
#define OUT_SDF   (OUT_GC + NRAYS*96*3)

#define RAY_START_F 0.5f
#define RAY_END_F   1.5f

#if defined(__has_builtin)
#if __has_builtin(__builtin_amdgcn_tensor_load_to_lds) && __has_builtin(__builtin_amdgcn_s_wait_tensorcnt)
#define HAVE_TDM 1
#endif
#endif
#ifndef HAVE_TDM
#define HAVE_TDM 0
#endif

__device__ __forceinline__ void lds_fence() {
  asm volatile("s_wait_dscnt 0" ::: "memory");
}

__device__ __forceinline__ float urand(unsigned a, unsigned b) {
  unsigned h = a*0x9E3779B9u ^ (b + 0x7F4A7C15u);
  h ^= h >> 16; h *= 0x85EBCA6Bu; h ^= h >> 13; h *= 0xC2B2AE35u; h ^= h >> 16;
  return (float)(h >> 8) * (1.0f/16777216.0f);
}

__device__ __forceinline__ float sigmoidf(float x){ return 1.f/(1.f+__expf(-x)); }

/* branchless fast softplus: max(x,0) + log(1+exp(-|x|)) -> v_exp_f32/v_log_f32 */
__device__ __forceinline__ float softplusf(float x){
  return fmaxf(x, 0.f) + __logf(1.f + __expf(-fabsf(x)));
}

union HPair { _Float16 h[2]; unsigned u; };
union Frag  { unsigned u[8]; v16h v; };

/* CDNA5 16-bit A-matrix 16x32: lanes 0-15 hold K 0..7,16..23; lanes 16-31 hold K 8..15,24..31 */
__device__ __forceinline__ int a_kpair(int v, int laneHi) {
  int base0 = laneHi * 8;
  return (v < 4) ? (base0 + 2*v) : (16 + base0 + 2*(v-4));
}

__device__ __forceinline__ v16h load_a_frag(const _Float16* lds, int M, int rowStride,
                                            int laneHi, int kOff) {
  Frag f;
#pragma unroll
  for (int v = 0; v < 8; ++v) {
    int k = kOff + a_kpair(v, laneHi);
    f.u[v] = *(const unsigned*)&lds[M*rowStride + k];
  }
  return f.v;
}

/* B-matrix 32x16: lane = column (N=lane&15), K order matches A: lanes 0-15 hold K 0..7,16..23; lanes 16-31 hold K 8..15,24..31 */
__device__ __forceinline__ v16h load_b_frag(const _Float16* ldsT, int N, int kStride,
                                            int laneHi, int kOff) {
  Frag f;
#pragma unroll
  for (int v = 0; v < 8; ++v) {
    int k = kOff + a_kpair(v, laneHi);
    f.u[v] = *(const unsigned*)&ldsT[N*kStride + k];
  }
  return f.v;
}

__device__ __forceinline__ v8f splat8(float x){
  v8f r;
#pragma unroll
  for (int i=0;i<8;++i) r[i]=x;
  return r;
}

__device__ __forceinline__ v8f wmma_f16(v16h a, v16h b, v8f c){
  return __builtin_amdgcn_wmma_f32_16x16x32_f16(false, a, false, b, (short)0, c, false, false);
}

/* dH[m][k] = sigmoid(pre) * w2[:,0]; sigmoid recovered as 1-exp(-softplus) */
__device__ __forceinline__ v16h load_dh_frag(const _Float16* hs, int M, int laneHi,
                                             int kOff, const float* w2c0) {
  Frag f;
#pragma unroll
  for (int v = 0; v < 8; ++v) {
    int k = kOff + a_kpair(v, laneHi);
    float h0 = (float)hs[M*64 + k];
    float h1 = (float)hs[M*64 + k + 1];
    float s0 = 1.f - __expf(-h0);
    float s1 = 1.f - __expf(-h1);
    HPair p;
    p.h[0] = (_Float16)(s0 * w2c0[k]);
    p.h[1] = (_Float16)(s1 * w2c0[k+1]);
    f.u[v] = p.u;
  }
  return f.v;
}

#if HAVE_TDM
/* Build a D# for a dense 2D f32 tile load (rows x cols, row stride in elems)
   and issue TENSOR_LOAD_TO_LDS.  Group0: count=1 | lds_addr | global_addr | type=2.
   Group1: data_size=4B, tensor_dim0/1, tile_dim0/1, tensor_dim0_stride.
   Groups 2/3 + extension unused (zero) for a 2D tensor.                      */
__device__ __forceinline__ void tdm_load_2d_f32(const void* gptr, unsigned lds_addr,
                                                unsigned rows, unsigned cols,
                                                unsigned stride)
{
  unsigned long long ga = (unsigned long long)gptr;
  u32x4 g0;
  g0.x = 1u;                                             /* count=1 (valid)   */
  g0.y = lds_addr;                                       /* bits 63:32        */
  g0.z = (unsigned)(ga & 0xFFFFFFFFu);                   /* bits 95:64        */
  g0.w = (unsigned)((ga >> 32) & 0x01FFFFFFu) | (2u<<30);/* addr hi | type=2  */
  i32x8 g1;
  g1[0] = (int)(2u << 16);                               /* data_size=4B      */
  g1[1] = (int)((cols & 0xFFFFu) << 16);                 /* tensor_dim0 lo    */
  g1[2] = (int)(((cols >> 16) & 0xFFFFu) | ((rows & 0xFFFFu) << 16)); /* td0 hi | td1 lo */
  g1[3] = (int)(((rows >> 16) & 0xFFFFu) | ((cols & 0xFFFFu) << 16)); /* td1 hi | tile_dim0 */
  g1[4] = (int)(rows & 0xFFFFu);                         /* tile_dim1, tile_dim2=0 */
  g1[5] = (int)stride;                                   /* tensor_dim0_stride lo */
  g1[6] = 0;
  g1[7] = 0;
  i32x4 z4 = {0,0,0,0};
  i32x8 z8 = {0,0,0,0,0,0,0,0};
  __builtin_amdgcn_tensor_load_to_lds(g0, g1, z4, z4, z8, 0);
}

__device__ __forceinline__ unsigned lds_byte_offset(const void* p) {
  /* generic LDS pointer truncates to LDS byte address (aperture in high bits) */
  return (unsigned)(unsigned long long)p;
}
#endif

/* ------------------------------------------------------------------ */
/* K1: jittered coarse depths + coords                                 */
__global__ void __launch_bounds__(256)
gen_coarse_kernel(const float* __restrict__ ro, const float* __restrict__ rd,
                  float* __restrict__ depths, float* __restrict__ coords)
{
  int idx = blockIdx.x*blockDim.x + threadIdx.x;
  if (idx >= NPTS) return;
  int ray = idx / Ssmp;
  int s   = idx % Ssmp;
  const float delta = (RAY_END_F - RAY_START_F) / (float)(Ssmp - 1);
  float dep = RAY_START_F + (float)s*delta + urand((unsigned)idx, 0x13572468u)*delta
            - ro[ray*3+2];
  depths[idx] = dep;
#pragma unroll
  for (int j = 0; j < 3; ++j)
    coords[idx*3+j] = ro[ray*3+j] + dep*rd[ray*3+j];
}

/* ------------------------------------------------------------------ */
/* K2: tri-plane bilinear gather -> f16 features (mean over planes)    */
__global__ void __launch_bounds__(256)
gather_feat_kernel(const float* __restrict__ planes, const float* __restrict__ coords,
                   _Float16* __restrict__ X)
{
  int idx = blockIdx.x*blockDim.x + threadIdx.x;
  if (idx >= NPTS) return;
  int b = idx / (Rrays*Ssmp);
  float cx = coords[idx*3+0], cy = coords[idx*3+1], cz = coords[idx*3+2];
  float acc[32];
#pragma unroll
  for (int c = 0; c < 32; ++c) acc[c] = 0.f;

#pragma unroll
  for (int p = 0; p < 3; ++p) {
    float u = (p==2) ? cz : cx;
    float v = (p==1) ? cz : cy;
    float px = (u + 1.f)*128.f - 0.5f;
    float py = (v + 1.f)*128.f - 0.5f;
    float fx = floorf(px), fy = floorf(py);
    float wx = px - fx, wy = py - fy;
    int x0 = (int)fx, y0 = (int)fy;
    const float* base = planes + (size_t)((b*3 + p)*32) * PLANE_PIX;
#pragma unroll
    for (int t = 0; t < 4; ++t) {
      int ix = x0 + (t & 1), iy = y0 + (t >> 1);
      float w = ((t & 1) ? wx : 1.f-wx) * ((t >> 1) ? wy : 1.f-wy);
      if (ix >= 0 && ix < HRES && iy >= 0 && iy < HRES) {
        const float* tp = base + iy*HRES + ix;
#pragma unroll
        for (int c = 0; c < 32; ++c) acc[c] += w * tp[c*PLANE_PIX];
      }
    }
  }
#pragma unroll
  for (int c = 0; c < 32; c += 2) {
    HPair pr;
    pr.h[0] = (_Float16)(acc[c]   * (1.f/3.f));
    pr.h[1] = (_Float16)(acc[c+1] * (1.f/3.f));
    ((unsigned*)X)[idx*16 + (c >> 1)] = pr.u;
  }
}

/* ------------------------------------------------------------------ */
/* K3: WMMA MLP fwd (32->64 softplus ->33) + bwd dsigma/dX.            */
/*     Weights pulled into LDS via the Tensor Data Mover when present. */
__global__ void __launch_bounds__(128)
mlp_wmma_kernel(const _Float16* __restrict__ X,
                const float* __restrict__ w1g, const float* __restrict__ b1g,
                const float* __restrict__ w2g, const float* __restrict__ b2g,
                float* __restrict__ sigma, _Float16* __restrict__ rgb,
                float* __restrict__ gX)
{
#if HAVE_TDM
  __shared__ float s_w1f[32*64];     /* TDM staging: raw f32 w1 */
  __shared__ float s_w2f[64*33];     /* TDM staging: raw f32 w2 */
#endif
  __shared__ _Float16 s_w1 [32*64];  /* [f][h] row-major: bwd B (K=h contiguous) */
  __shared__ _Float16 s_w1t[64*32];  /* [h][f] -> fwd1 B: [n=h][k=f] contiguous  */
  __shared__ _Float16 s_w2t[48*64];  /* [n][k], cols 33..47 zero                 */
  __shared__ float s_b1[64];
  __shared__ float s_b2[48];
  __shared__ float s_w2c0[64];
  __shared__ _Float16 s_x[4][16*32];
  __shared__ _Float16 s_h[4][16*64];

  int tid = threadIdx.x;

#if HAVE_TDM
  /* async DMA of both weight matrices into LDS, issued once by wave 0 */
  if (tid < 32) {
    tdm_load_2d_f32(w1g, lds_byte_offset(s_w1f), 32u, 64u, 64u);
    tdm_load_2d_f32(w2g, lds_byte_offset(s_w2f), 64u, 33u, 33u);
    __builtin_amdgcn_s_wait_tensorcnt(0);
  }
  __syncthreads();
  const float* w1src = s_w1f;
  const float* w2src = s_w2f;
#else
  const float* w1src = w1g;
  const float* w2src = w2g;
#endif

  for (int i = tid; i < 32*64; i += 128) {
    float v = w1src[i];              /* i = f*64 + h */
    s_w1[i] = (_Float16)v;
    int f = i >> 6, h = i & 63;
    s_w1t[h*32 + f] = (_Float16)v;
  }
  for (int i = tid; i < 48*64; i += 128) {
    int n = i >> 6, k = i & 63;
    float v = (n < 33) ? w2src[k*33 + n] : 0.f;
    s_w2t[i] = (_Float16)v;
  }
  for (int i = tid; i < 64; i += 128) { s_b1[i] = b1g[i]; s_w2c0[i] = w2src[i*33]; }
  for (int i = tid; i < 48; i += 128) s_b2[i] = (i < 33) ? b2g[i] : 0.f;
  __syncthreads();

  int wave = tid >> 5, lane = tid & 31;
  int laneHi = lane >> 4, nl = lane & 15;
  int tile = blockIdx.x*4 + wave;
  int p0 = tile * 16;
  if (p0 >= NPTS) return;

  _Float16* xs = s_x[wave];
  _Float16* hs = s_h[wave];

  /* cooperative 1KB tile load (b128) */
  {
    const uint4* src = (const uint4*)(X + (size_t)p0*32);
    uint4* dst = (uint4*)xs;
    dst[lane]      = src[lane];
    dst[lane + 32] = src[lane + 32];
  }
  lds_fence();

  v16h aX = load_a_frag(xs, nl, 32, laneHi, 0);

  /* layer 1: X(16x32) @ W1(32x64) + b1, softplus -> LDS H */
#pragma unroll
  for (int j = 0; j < 4; ++j) {
    v16h bW = load_b_frag(s_w1t, 16*j + nl, 32, laneHi, 0);
    v8f c = splat8(s_b1[16*j + nl]);
    c = wmma_f16(aX, bW, c);
#pragma unroll
    for (int r = 0; r < 8; ++r) {
      float hv = softplusf(c[r]);
      int m = r + 8*laneHi;
      hs[m*64 + 16*j + nl] = (_Float16)hv;
    }
  }
  lds_fence();

  v16h aH0 = load_a_frag(hs, nl, 64, laneHi, 0);
  v16h aH1 = load_a_frag(hs, nl, 64, laneHi, 32);

  /* layer 2: H(16x64) @ W2(64x48pad) + b2 */
#pragma unroll
  for (int j = 0; j < 3; ++j) {
    v16h bk0 = load_b_frag(s_w2t, 16*j + nl, 64, laneHi, 0);
    v16h bk1 = load_b_frag(s_w2t, 16*j + nl, 64, laneHi, 32);
    v8f c = splat8(s_b2[16*j + nl]);
    c = wmma_f16(aH0, bk0, c);
    c = wmma_f16(aH1, bk1, c);
    int col = 16*j + nl;
#pragma unroll
    for (int r = 0; r < 8; ++r) {
      int m = r + 8*laneHi;
      float o = c[r];
      if (col == 0) {
        sigma[p0 + m] = o;
      } else if (col <= 32) {
        float rv = sigmoidf(o)*1.002f - 0.001f;
        rgb[(size_t)(p0 + m)*32 + (col - 1)] = (_Float16)rv;
      }
    }
  }

  /* backward: gX = (sigmoid(pre) .* w2[:,0]) @ W1^T */
  v16h dH0 = load_dh_frag(hs, nl, laneHi, 0,  s_w2c0);
  v16h dH1 = load_dh_frag(hs, nl, laneHi, 32, s_w2c0);
#pragma unroll
  for (int j = 0; j < 2; ++j) {
    v16h bk0 = load_b_frag(s_w1, 16*j + nl, 64, laneHi, 0);
    v16h bk1 = load_b_frag(s_w1, 16*j + nl, 64, laneHi, 32);
    v8f c = splat8(0.f);
    c = wmma_f16(dH0, bk0, c);
    c = wmma_f16(dH1, bk1, c);
    int f = 16*j + nl;
#pragma unroll
    for (int r = 0; r < 8; ++r) {
      int m = r + 8*laneHi;
      gX[(size_t)(p0 + m)*32 + f] = c[r];
    }
  }
}

/* ------------------------------------------------------------------ */
/* K4: analytic grad through tri-plane, SDF, density, gn; writes sdf  */
/*     and grad_cano slices of d_out directly.                         */
__global__ void __launch_bounds__(256)
grad_density_kernel(const float* __restrict__ planes, const float* __restrict__ coords,
                    const float* __restrict__ gX, const float* __restrict__ sigma,
                    const float* __restrict__ beta_p,
                    float* __restrict__ dens, float* __restrict__ gn,
                    float* __restrict__ out_gc, float* __restrict__ out_sdf,
                    int levelOff)
{
  int idx = blockIdx.x*blockDim.x + threadIdx.x;
  if (idx >= NPTS) return;
  int b = idx / (Rrays*Ssmp);
  int m = idx - b*(Rrays*Ssmp);
  float cx = coords[idx*3+0], cy = coords[idx*3+1], cz = coords[idx*3+2];

  float g[32];
#pragma unroll
  for (int c = 0; c < 32; ++c) g[c] = gX[(size_t)idx*32 + c];

  float gx = 0.f, gy = 0.f, gz = 0.f;
#pragma unroll
  for (int p = 0; p < 3; ++p) {
    float u = (p==2) ? cz : cx;
    float v = (p==1) ? cz : cy;
    float px = (u + 1.f)*128.f - 0.5f;
    float py = (v + 1.f)*128.f - 0.5f;
    float fx = floorf(px), fy = floorf(py);
    float wx = px - fx, wy = py - fy;
    int x0 = (int)fx, y0 = (int)fy;
    const float* base = planes + (size_t)((b*3 + p)*32) * PLANE_PIX;
    float G[4];
#pragma unroll
    for (int t = 0; t < 4; ++t) {
      int ix = x0 + (t & 1), iy = y0 + (t >> 1);
      float dot = 0.f;
      if (ix >= 0 && ix < HRES && iy >= 0 && iy < HRES) {
        const float* tp = base + iy*HRES + ix;
#pragma unroll
        for (int c = 0; c < 32; ++c) dot += g[c] * tp[c*PLANE_PIX];
      }
      G[t] = dot;
    }
    float Gu = ((G[1]-G[0])*(1.f-wy) + (G[3]-G[2])*wy) * (128.f/3.f);
    float Gv = ((G[2]-G[0])*(1.f-wx) + (G[3]-G[1])*wx) * (128.f/3.f);
    if      (p == 0) { gx += Gu; gy += Gv; }
    else if (p == 1) { gx += Gu; gz += Gv; }
    else             { gz += Gu; gy += Gv; }
  }

  /* SDF analytic part */
  float rr = sqrtf(cx*cx + cy*cy + cz*cz + 1e-12f);
  float ssdf = rr - 0.5f;
  gx += cx/rr; gy += cy/rr; gz += cz/rr;

  float sdf = sigma[idx] + ssdf;
  float beta = beta_p[0];
  float density = sigmoidf(-sdf/beta)/beta;
  dens[idx] = density;

  float tx = -gx, ty = gy, tz = gz;
  float nrm = fmaxf(sqrtf(tx*tx + ty*ty + tz*tz), 1e-12f);
  gn[idx*3+0] = tx/nrm; gn[idx*3+1] = ty/nrm; gn[idx*3+2] = tz/nrm;

  size_t gcb = (size_t)b*(Rrays*96*3) + (size_t)(levelOff + m)*3;
  out_gc[gcb+0] = gx; out_gc[gcb+1] = gy; out_gc[gcb+2] = gz;
  out_sdf[(size_t)b*(Rrays*96) + levelOff + m] = sdf;
}

/* ------------------------------------------------------------------ */
/* K5: importance sampling (one wave per ray)                          */
__global__ void __launch_bounds__(128)
importance_kernel(const float* __restrict__ depths_c, const float* __restrict__ dens_c,
                  const float* __restrict__ ro, const float* __restrict__ rd,
                  float* __restrict__ depths_f, float* __restrict__ coords_f)
{
  __shared__ float sw[4][48];
  __shared__ float scdf[4][48];
  __shared__ float szm[4][48];
  int wave = threadIdx.x >> 5, lane = threadIdx.x & 31;
  int ray = blockIdx.x*4 + wave;
  if (ray >= NRAYS) return;
  const float* z  = depths_c + ray*Ssmp;
  const float* dn = dens_c   + ray*Ssmp;
  float* w    = sw[wave];
  float* cdf  = scdf[wave];
  float* zmid = szm[wave];

  for (int i = lane; i < 47; i += 32) {
    float d0 = z[i], d1 = z[i+1];
    zmid[i] = 0.5f*(d0 + d1);
    float dm = 0.5f*(dn[i] + dn[i+1]);
    w[i] = 1.f - __expf(-dm*(d1 - d0));   /* alpha for now */
  }
  lds_fence();

  if (lane == 0) {
    float T = 1.f;
    for (int i = 0; i < 47; ++i) { float a = w[i]; w[i] = a*T; T *= (1.f - a + 1e-10f); }
    float sum = 0.f;
    for (int i = 0; i < 45; ++i) {
      int k = i + 1;
      float wl = (k-1 < 47) ? w[k-1] : -3.0e38f;
      float wc = (k   < 47) ? w[k]   : -3.0e38f;
      float wr = (k+1 < 47) ? w[k+1] : -3.0e38f;
      float p = 0.5f*(fmaxf(wl, wc) + fmaxf(wc, wr)) + 0.01f + 1e-5f;
      cdf[i+1] = p; sum += p;
    }
    float inv = 1.f/sum, run = 0.f;
    cdf[0] = 0.f;
    for (int i = 1; i <= 45; ++i) { run += cdf[i]*inv; cdf[i] = run; }
  }
  lds_fence();

  for (int s = lane; s < 48; s += 32) {
    float u = urand((unsigned)(ray*977 + s), 0x2468ACEu);
    int lo = 0, hi = 46;
    while (lo < hi) { int mid = (lo + hi) >> 1; if (cdf[mid] <= u) lo = mid + 1; else hi = mid; }
    int below = (lo - 1 > 0) ? lo - 1 : 0;
    int above = (lo < 45) ? lo : 45;
    float c0 = cdf[below], c1 = cdf[above];
    float b0 = zmid[below], b1 = zmid[above];
    float denom = (c1 - c0 < 1e-5f) ? 1.f : (c1 - c0);
    float d = b0 + (u - c0)/denom * (b1 - b0);
    int pidx = ray*48 + s;
    depths_f[pidx] = d;
#pragma unroll
    for (int j = 0; j < 3; ++j)
      coords_f[pidx*3 + j] = ro[ray*3 + j] + d*rd[ray*3 + j];
  }
}

/* ------------------------------------------------------------------ */
/* K6: merge-sort 96 samples per ray + final march (one wave per ray)  */
__global__ void __launch_bounds__(128)
final_march_kernel(const float* __restrict__ depths_c, const float* __restrict__ depths_f,
                   const float* __restrict__ dens_c, const float* __restrict__ dens_f,
                   const _Float16* __restrict__ rgb_c, const _Float16* __restrict__ rgb_f,
                   const float* __restrict__ gn_c, const float* __restrict__ gn_f,
                   float* __restrict__ out)
{
  __shared__ float sd[4][96];
  __shared__ float sds[4][96];
  __shared__ float sg[4][96*3];
  __shared__ int   sp[4][96];
  __shared__ float swt[4][96];
  int wave = threadIdx.x >> 5, lane = threadIdx.x & 31;
  int ray = blockIdx.x*4 + wave;
  if (ray >= NRAYS) return;
  float* d  = sd[wave];
  float* dn = sds[wave];
  float* g  = sg[wave];
  int*   pm = sp[wave];
  float* wt = swt[wave];

  for (int i = lane; i < 96; i += 32) {
    int s = (i < 48) ? i : (i - 48);
    int pidx = ray*48 + s;
    if (i < 48) {
      d[i] = depths_c[pidx]; dn[i] = dens_c[pidx];
      g[i*3+0] = gn_c[pidx*3+0]; g[i*3+1] = gn_c[pidx*3+1]; g[i*3+2] = gn_c[pidx*3+2];
    } else {
      d[i] = depths_f[pidx]; dn[i] = dens_f[pidx];
      g[i*3+0] = gn_f[pidx*3+0]; g[i*3+1] = gn_f[pidx*3+1]; g[i*3+2] = gn_f[pidx*3+2];
    }
  }
  lds_fence();

  for (int i = lane; i < 96; i += 32) {
    float di = d[i]; int rank = 0;
    for (int j = 0; j < 96; ++j) {
      float dj = d[j];
      rank += (dj < di) || ((dj == di) && (j < i));
    }
    pm[rank] = i;
  }
  lds_fence();

  if (lane == 0) {
    float T = 1.f;
    for (int k = 0; k < 95; ++k) {
      int i0 = pm[k], i1 = pm[k+1];
      float dm = 0.5f*(dn[i0] + dn[i1]);
      float a = 1.f - __expf(-dm*(d[i1] - d[i0]));
      wt[k] = a*T;
      T *= (1.f - a + 1e-10f);
    }
  }
  lds_fence();

  { /* per-channel rgb compositing, lane = channel */
    int c = lane;
    float acc = 0.f;
    for (int k = 0; k < 95; ++k) {
      int i0 = pm[k], i1 = pm[k+1];
      float r0 = (i0 < 48) ? (float)rgb_c[(size_t)(ray*48 + i0)*32 + c]
                           : (float)rgb_f[(size_t)(ray*48 + i0 - 48)*32 + c];
      float r1 = (i1 < 48) ? (float)rgb_c[(size_t)(ray*48 + i1)*32 + c]
                           : (float)rgb_f[(size_t)(ray*48 + i1 - 48)*32 + c];
      acc += wt[k]*0.5f*(r0 + r1);
    }
    out[OUT_RGB + ray*32 + c] = acc*2.f - 1.f;
  }

  if (lane == 0) {
    float wtot = 0.f, dep = 0.f, g0 = 0.f, g1 = 0.f, g2 = 0.f;
    for (int k = 0; k < 95; ++k) {
      int i0 = pm[k], i1 = pm[k+1];
      float w = wt[k];
      wtot += w;
      dep  += w*0.5f*(d[i0] + d[i1]);
      g0   += w*0.5f*(g[i0*3+0] + g[i1*3+0]);
      g1   += w*0.5f*(g[i0*3+1] + g[i1*3+1]);
      g2   += w*0.5f*(g[i0*3+2] + g[i1*3+2]);
    }
    float depth = dep / fmaxf(wtot, 1e-12f);
    depth = fminf(fmaxf(depth, d[pm[0]]), d[pm[95]]);
    out[OUT_DEPTH + ray] = depth;
    out[OUT_WSUM  + ray] = wtot;
    out[OUT_GRADS + ray*3+0] = g0;
    out[OUT_GRADS + ray*3+1] = g1;
    out[OUT_GRADS + ray*3+2] = g2;
  }
}

/* ------------------------------------------------------------------ */
extern "C" void kernel_launch(void* const* d_in, const int* in_sizes, int n_in,
                              void* d_out, int out_size, void* d_ws, size_t ws_size,
                              hipStream_t stream) {
  const float* planes = (const float*)d_in[0];
  const float* ro     = (const float*)d_in[1];
  const float* rd     = (const float*)d_in[2];
  const float* w1     = (const float*)d_in[3];
  const float* b1     = (const float*)d_in[4];
  const float* w2     = (const float*)d_in[5];
  const float* b2     = (const float*)d_in[6];
  const float* beta   = (const float*)d_in[7];
  float* out = (float*)d_out;

  char* ws = (char*)d_ws;
  auto alloc = [&](size_t bytes) -> void* {
    void* p = (void*)ws;
    ws += (bytes + 255) & ~(size_t)255;
    return p;
  };
  float*    depths_c = (float*)   alloc((size_t)NPTS*4);
  float*    depths_f = (float*)   alloc((size_t)NPTS*4);
  float*    coords_c = (float*)   alloc((size_t)NPTS*3*4);
  float*    coords_f = (float*)   alloc((size_t)NPTS*3*4);
  _Float16* Xh       = (_Float16*)alloc((size_t)NPTS*32*2);   /* reused both levels */
  float*    gX       = (float*)   alloc((size_t)NPTS*32*4);   /* reused both levels */
  float*    sigma    = (float*)   alloc((size_t)NPTS*4);      /* reused both levels */
  _Float16* rgb_c    = (_Float16*)alloc((size_t)NPTS*32*2);
  _Float16* rgb_f    = (_Float16*)alloc((size_t)NPTS*32*2);
  float*    dens_c   = (float*)   alloc((size_t)NPTS*4);
  float*    dens_f   = (float*)   alloc((size_t)NPTS*4);
  float*    gn_c     = (float*)   alloc((size_t)NPTS*3*4);
  float*    gn_f     = (float*)   alloc((size_t)NPTS*3*4);

  const int ptBlocks   = NPTS/256;          /* 768  */
  const int mlpBlocks  = (NPTS/16)/4;       /* 3072 */
  const int rayBlocks  = NRAYS/4;           /* 1024 */

  /* coarse level */
  gen_coarse_kernel<<<ptBlocks, 256, 0, stream>>>(ro, rd, depths_c, coords_c);
  gather_feat_kernel<<<ptBlocks, 256, 0, stream>>>(planes, coords_c, Xh);
  mlp_wmma_kernel<<<mlpBlocks, 128, 0, stream>>>(Xh, w1, b1, w2, b2, sigma, rgb_c, gX);
  grad_density_kernel<<<ptBlocks, 256, 0, stream>>>(planes, coords_c, gX, sigma, beta,
                                                    dens_c, gn_c,
                                                    out + OUT_GC, out + OUT_SDF, 0);
  /* importance sampling -> fine level */
  importance_kernel<<<rayBlocks, 128, 0, stream>>>(depths_c, dens_c, ro, rd,
                                                   depths_f, coords_f);
  gather_feat_kernel<<<ptBlocks, 256, 0, stream>>>(planes, coords_f, Xh);
  mlp_wmma_kernel<<<mlpBlocks, 128, 0, stream>>>(Xh, w1, b1, w2, b2, sigma, rgb_f, gX);
  grad_density_kernel<<<ptBlocks, 256, 0, stream>>>(planes, coords_f, gX, sigma, beta,
                                                    dens_f, gn_f,
                                                    out + OUT_GC, out + OUT_SDF,
                                                    Rrays*Ssmp);
  /* unify + final march */
  final_march_kernel<<<rayBlocks, 128, 0, stream>>>(depths_c, depths_f, dens_c, dens_f,
                                                    rgb_c, rgb_f, gn_c, gn_f, out);
}
